// Attention_43963285242711
// MI455X (gfx1250) — matrix operand order repaired and checked by reading, not compiled, not run on hardware
//
#include <hip/hip_runtime.h>
#include <hip/hip_bf16.h>

typedef __attribute__((ext_vector_type(16))) _Float16 v16h;
typedef __attribute__((ext_vector_type(8)))  float    v8f;

// async-to-LDS builtin operand types (probe-revealed: v4i in AS1 / AS3)
typedef int vint4_t __attribute__((vector_size(16)));
typedef __attribute__((address_space(1))) vint4_t as1_vint4;
typedef __attribute__((address_space(3))) vint4_t as3_vint4;

#if defined(__has_builtin)
#  if __has_builtin(__builtin_amdgcn_global_load_async_to_lds_b128) && \
      __has_builtin(__builtin_amdgcn_s_wait_asynccnt)
#    define HAS_ASYNC_LDS 1
#  endif
#endif
#ifndef HAS_ASYNC_LDS
#  define HAS_ASYNC_LDS 0
#endif

#define L_SEQ   2048
#define D_MODEL 2048
#define NH      32
#define NKV     8
#define HD      64
#define GQA_REP (NH / NKV)   // 4

// GEMM block tiling
#define BM 128
#define BN 64
#define BK 32
#define XS_STRIDE 36   // padded fp32 row stride (144B, 16B aligned, bank-skewed)
#define WS_STRIDE 36

// ---------------------------------------------------------------------------
// Fragment layouts (CDNA5 wave32 WMMA, CDNA5 ISA §7.12.2)
//  A 16x32 f16 : M = lane&15 ; vgpr v, half h -> K = (v<4?0:16) +
//                (lane>=16?8:0) + 2*(v&3) + h   (two contiguous 8-runs)
//  B 32x16 f16 : N = lane&15 ; vgpr v, half h -> K = (v<4?0:16) + (lane>=16?8:0) + 2*(v&3) + h
//                (same two 8-runs as A)
//  C/D 16x16 f32: vgpr r -> M = r + (lane>=16?8:0) ; N = lane&15
// ---------------------------------------------------------------------------

__device__ inline v16h fragA_f32(const float* row, int hiA) {
  v16h f;
#pragma unroll
  for (int v = 0; v < 4; ++v) {
    f[2 * v]     = (_Float16)row[hiA + 2 * v];
    f[2 * v + 1] = (_Float16)row[hiA + 2 * v + 1];
    f[2 * v + 8] = (_Float16)row[16 + hiA + 2 * v];
    f[2 * v + 9] = (_Float16)row[16 + hiA + 2 * v + 1];
  }
  return f;
}

__device__ inline v16h fragB_f32(const float* row, int hiB) {
  v16h f;
#pragma unroll
  for (int i = 0; i < 16; ++i) f[i] = (_Float16)row[(i < 8 ? 0 : 16) + (hiB >> 1) + (i & 7)];
  return f;
}

// Y[M,N] = X[M,K] @ W[N,K]^T   (fp32 in, f16 WMMA compute, fp32 out)
// 256 threads = 8 waves; block tile 128x64, BK=32, double-buffered LDS fed by
// async global->LDS DMA (ASYNCcnt). Wave (wm,wn) owns a 32x32 sub-tile:
// 2 A-frags x 2 B-frags -> 4 WMMAs per k-step, fragments reused.
__global__ __launch_bounds__(256)
void gemm_xwT(const float* __restrict__ X, const float* __restrict__ W,
              float* __restrict__ Y, int M, int N, int K) {
  __shared__ __align__(16) float Xs[2][BM * XS_STRIDE];
  __shared__ __align__(16) float Ws[2][BN * WS_STRIDE];
  (void)M;

  const int tid  = threadIdx.x;
  const int lane = tid & 31;
  const int wave = tid >> 5;          // 0..7
  const int wm   = wave & 3;          // M sub-tile (4 x 32 rows)
  const int wn   = wave >> 2;         // N sub-tile (2 x 32 cols)
  const int m0   = blockIdx.y * BM;
  const int n0   = blockIdx.x * BN;

  const int l15 = lane & 15;
  const int hiA = (lane & 16) ? 8 : 0;
  const int hiB = (lane & 16) ? 16 : 0;
  const int hi8 = (lane & 16) ? 8 : 0;

  // staging map: X tile 128 rows x 32 floats -> 256 thr x (1 row-half = 16 floats)
  const int xrow = tid >> 1;
  const int xcol = (tid & 1) * 16;
  // W tile 64 rows x 32 floats -> 256 thr x (1 row-quarter = 8 floats)
  const int wrow = tid >> 2;
  const int wcol = (tid & 3) * 8;

  auto stage = [&](int k0, int buf) {
    const float* xsrc = X + (size_t)(m0 + xrow) * K + k0 + xcol;
    const float* wsrc = W + (size_t)(n0 + wrow) * K + k0 + wcol;
    float* xdst = &Xs[buf][xrow * XS_STRIDE + xcol];
    float* wdst = &Ws[buf][wrow * WS_STRIDE + wcol];
#if HAS_ASYNC_LDS
#pragma unroll
    for (int i = 0; i < 4; ++i)
      __builtin_amdgcn_global_load_async_to_lds_b128(
          (as1_vint4*)(void*)(xsrc + 4 * i), (as3_vint4*)(void*)(xdst + 4 * i), 0, 0);
#pragma unroll
    for (int i = 0; i < 2; ++i)
      __builtin_amdgcn_global_load_async_to_lds_b128(
          (as1_vint4*)(void*)(wsrc + 4 * i), (as3_vint4*)(void*)(wdst + 4 * i), 0, 0);
#else
#pragma unroll
    for (int i = 0; i < 4; ++i)
      *(float4*)(xdst + 4 * i) = *(const float4*)(xsrc + 4 * i);
#pragma unroll
    for (int i = 0; i < 2; ++i)
      *(float4*)(wdst + 4 * i) = *(const float4*)(wsrc + 4 * i);
#endif
  };

  v8f acc[2][2] = {{v8f{}, v8f{}}, {v8f{}, v8f{}}};

  const int nk = K / BK;
  stage(0, 0);
  for (int i = 0; i < nk; ++i) {
    const int buf = i & 1;
    if (i + 1 < nk) {
      stage((i + 1) * BK, buf ^ 1);      // prefetch next tile into other buffer
#if HAS_ASYNC_LDS
      __builtin_amdgcn_s_wait_asynccnt(6);   // 6 outstanding = next tile only
#endif
    } else {
#if HAS_ASYNC_LDS
      __builtin_amdgcn_s_wait_asynccnt(0);
#endif
    }
    __syncthreads();

    v16h afr[2], bfr[2];
    afr[0] = fragA_f32(&Xs[buf][(wm * 32 + l15) * XS_STRIDE], hiA);
    afr[1] = fragA_f32(&Xs[buf][(wm * 32 + 16 + l15) * XS_STRIDE], hiA);
    bfr[0] = fragB_f32(&Ws[buf][(wn * 32 + l15) * WS_STRIDE], hiB);
    bfr[1] = fragB_f32(&Ws[buf][(wn * 32 + 16 + l15) * WS_STRIDE], hiB);
#pragma unroll
    for (int ti = 0; ti < 2; ++ti)
#pragma unroll
      for (int tj = 0; tj < 2; ++tj)
        acc[ti][tj] = __builtin_amdgcn_wmma_f32_16x16x32_f16(
            false, afr[ti], false, bfr[tj], (short)0, acc[ti][tj], false, false);

    __syncthreads();   // protect LDS buffer before it is overwritten
  }

#pragma unroll
  for (int ti = 0; ti < 2; ++ti) {
    const int row0 = m0 + wm * 32 + ti * 16 + hi8;
#pragma unroll
    for (int tj = 0; tj < 2; ++tj) {
      const int col = n0 + wn * 32 + tj * 16 + l15;
#pragma unroll
      for (int r = 0; r < 8; ++r)
        Y[(size_t)(row0 + r) * N + col] = acc[ti][tj][r];
    }
  }
}

// RoPE (half-split / rotate-half) + fp32->f16 + repack [L, nh*HD] -> [nh][L][HD].
// For Q: extra = attn_scale * HD^-0.5 folded in. For K: extra = 1, scale_ptr null.
__global__ void rope_cvt(const float* __restrict__ src, _Float16* __restrict__ dst,
                         const float* __restrict__ scale_ptr, int nheads, float extra) {
  const int tid = blockIdx.x * blockDim.x + threadIdx.x;
  const int total = L_SEQ * nheads * (HD / 2);
  if (tid >= total) return;
  const int p = tid & 31;                 // pair index 0..31
  const int h = (tid >> 5) % nheads;
  const int l = tid / (nheads * 32);
  const float sc = extra * (scale_ptr ? scale_ptr[0] : 1.0f);
  // inv_freq = 10000^(-p/32) = exp(-p * ln(10000)/32)
  const float inv_freq = __expf((float)p * (-9.210340371976184f / 32.0f));
  float s, c;
  __sincosf((float)l * inv_freq, &s, &c);
  const int D = nheads * HD;
  const float t1 = src[(size_t)l * D + h * HD + p];
  const float t2 = src[(size_t)l * D + h * HD + p + 32];
  const size_t o = ((size_t)h * L_SEQ + l) * HD + p;
  dst[o]      = (_Float16)((t1 * c - t2 * s) * sc);
  dst[o + 32] = (_Float16)((t1 * s + t2 * c) * sc);
}

// Plain convert + repack for V.
__global__ void cvt_v(const float* __restrict__ src, _Float16* __restrict__ dst) {
  const int tid = blockIdx.x * blockDim.x + threadIdx.x;
  const int total = L_SEQ * NKV * HD;
  if (tid >= total) return;
  const int d = tid & (HD - 1);
  const int h = (tid >> 6) % NKV;
  const int l = tid / (NKV * HD);
  dst[((size_t)h * L_SEQ + l) * HD + d] =
      (_Float16)src[(size_t)l * (NKV * HD) + h * HD + d];
}

// Flash-attention forward: one wave = 16 query rows of one head.
// Qh already carries attn_scale * HD^-0.5. Causal key loop stops at diagonal.
__global__ __launch_bounds__(32)
void attn_fwd(const _Float16* __restrict__ Qh, const _Float16* __restrict__ Kh,
              const _Float16* __restrict__ Vh, float* __restrict__ Ob) {
  __shared__ _Float16 pbuf[16 * 32];   // P tile bounce: C-layout -> A-layout

  const int lane = threadIdx.x & 31;
  const int q0  = blockIdx.x * 16;
  const int h   = blockIdx.y;
  const int kvh = h / GQA_REP;

  const _Float16* Qp = Qh + ((size_t)h * L_SEQ + q0) * HD;
  const _Float16* Kp = Kh + (size_t)kvh * L_SEQ * HD;
  const _Float16* Vp = Vh + (size_t)kvh * L_SEQ * HD;

  const int mrow = lane & 15;
  const int hi8  = (lane & 16) ? 8 : 0;
  const int hiB  = (lane & 16) ? 16 : 0;

  // Q A-fragments: qa[j] covers d = 32j .. 32j+31
  v16h qa[2];
#pragma unroll
  for (int j = 0; j < 2; ++j) {
    const _Float16* qr = Qp + mrow * HD + j * 32;
#pragma unroll
    for (int v = 0; v < 8; ++v) {
      const int k = (v < 4 ? 0 : 16) + hi8 + 2 * (v & 3);
      qa[j][2 * v]     = qr[k];
      qa[j][2 * v + 1] = qr[k + 1];
    }
  }

  float mstate[8], lstate[8];
  v8f o[4] = {v8f{}, v8f{}, v8f{}, v8f{}};
#pragma unroll
  for (int r = 0; r < 8; ++r) { mstate[r] = -3.0e38f; lstate[r] = 0.0f; }

  const int kend = q0 + 16;            // exclusive causal bound on keys
  for (int kc = 0; kc < kend; kc += 32) {
    // ---- scores: two 16x16 tiles, K-dim = 64 (2 WMMAs each) ----
    v8f s[2] = {v8f{}, v8f{}};
#pragma unroll
    for (int t = 0; t < 2; ++t) {
      const int key = kc + t * 16 + (lane & 15);
      const _Float16* kr = Kp + (size_t)key * HD;
#pragma unroll
      for (int j = 0; j < 2; ++j) {
        v16h kb;
#pragma unroll
        for (int v = 0; v < 8; ++v) {
          const int kk = j * 32 + (v < 4 ? 0 : 16) + hi8 + 2 * (v & 3);
          kb[2 * v]     = kr[kk];
          kb[2 * v + 1] = kr[kk + 1];
        }
        s[t] = __builtin_amdgcn_wmma_f32_16x16x32_f16(false, qa[j], false, kb,
                                                      (short)0, s[t], false, false);
      }
    }

    // ---- causal mask + online softmax (row reductions in half-wave) ----
    float pnew[2][8];
#pragma unroll
    for (int r = 0; r < 8; ++r) {
      const int qrow = q0 + r + hi8;
      float rowmax = -3.0e38f;
#pragma unroll
      for (int t = 0; t < 2; ++t) {
        const int key = kc + t * 16 + (lane & 15);
        const float sv = (key <= qrow) ? s[t][r] : -3.0e38f;
        s[t][r] = sv;
        rowmax = fmaxf(rowmax, sv);
      }
      for (int off = 1; off < 16; off <<= 1)
        rowmax = fmaxf(rowmax, __shfl_xor(rowmax, off, 32));
      const float mnew = fmaxf(mstate[r], rowmax);
      const float corr = __expf(mstate[r] - mnew);
      float psum = 0.0f;
#pragma unroll
      for (int t = 0; t < 2; ++t) {
        const float p = __expf(s[t][r] - mnew);
        pnew[t][r] = p;
        psum += p;
      }
      for (int off = 1; off < 16; off <<= 1)
        psum += __shfl_xor(psum, off, 32);
      lstate[r] = lstate[r] * corr + psum;
      mstate[r] = mnew;
#pragma unroll
      for (int t4 = 0; t4 < 4; ++t4) o[t4][r] *= corr;
    }

    // ---- P: C-layout -> LDS -> A-layout (16x32 f16) ----
#pragma unroll
    for (int r = 0; r < 8; ++r) {
      const int row = r + hi8;
      pbuf[row * 32 + (lane & 15)]      = (_Float16)pnew[0][r];
      pbuf[row * 32 + 16 + (lane & 15)] = (_Float16)pnew[1][r];
    }
    __syncthreads();
    v16h pa;
    {
      const _Float16* pr = pbuf + mrow * 32;
#pragma unroll
      for (int v = 0; v < 8; ++v) {
        const int k = (v < 4 ? 0 : 16) + hi8 + 2 * (v & 3);
        pa[2 * v]     = pr[k];
        pa[2 * v + 1] = pr[k + 1];
      }
    }
    __syncthreads();

    // ---- O += P @ V : 4 n-tiles over d=64, K-dim = 32 keys ----
#pragma unroll
    for (int t = 0; t < 4; ++t) {
      v16h vb;
#pragma unroll
      for (int v = 0; v < 8; ++v) {
        const int kk = (v < 4 ? 0 : 16) + hi8 + 2 * (v & 3);   // key within chunk
        const _Float16* vr = Vp + (size_t)(kc + kk) * HD + t * 16 + (lane & 15);
        vb[2 * v]     = vr[0];
        vb[2 * v + 1] = vr[HD];
      }
      o[t] = __builtin_amdgcn_wmma_f32_16x16x32_f16(false, pa, false, vb,
                                                    (short)0, o[t], false, false);
    }
  }

  // ---- epilogue: normalize, write [L, NH*HD] fp32 for output projection ----
#pragma unroll
  for (int r = 0; r < 8; ++r) {
    const float inv = 1.0f / lstate[r];
    const int row = q0 + r + hi8;
#pragma unroll
    for (int t = 0; t < 4; ++t)
      Ob[(size_t)row * D_MODEL + h * HD + t * 16 + (lane & 15)] = o[t][r] * inv;
  }
}

extern "C" void kernel_launch(void* const* d_in, const int* in_sizes, int n_in,
                              void* d_out, int out_size, void* d_ws, size_t ws_size,
                              hipStream_t stream) {
  (void)in_sizes; (void)n_in; (void)out_size; (void)ws_size;
  const float* x   = (const float*)d_in[0];
  const float* asc = (const float*)d_in[1];
  // d_in[2]: additive causal mask — regenerated analytically in attn_fwd
  const float* Wq = (const float*)d_in[3];
  const float* Wk = (const float*)d_in[4];
  const float* Wv = (const float*)d_in[5];
  const float* Wo = (const float*)d_in[6];
  float* out = (float*)d_out;

  char* ws = (char*)d_ws;                               // 52 MB used
  float*    Qp = (float*)(ws);                          // [L, 2048] 16 MB
  float*    Kp = (float*)(ws + (16u << 20));            // [L,  512]  4 MB
  float*    Vp = (float*)(ws + (20u << 20));            // [L,  512]  4 MB
  _Float16* Qh = (_Float16*)(ws + (24u << 20));         // [32][L][64] 8 MB
  _Float16* Kh = (_Float16*)(ws + (32u << 20));         // [8][L][64]  2 MB
  _Float16* Vh = (_Float16*)(ws + (34u << 20));         // [8][L][64]  2 MB
  float*    Ob = (float*)(ws + (36u << 20));            // [L, 2048] 16 MB

  const dim3 gblk(256);
  gemm_xwT<<<dim3(D_MODEL / BN, L_SEQ / BM), gblk, 0, stream>>>(x, Wq, Qp, L_SEQ, D_MODEL, D_MODEL);
  gemm_xwT<<<dim3((NKV * HD) / BN, L_SEQ / BM), gblk, 0, stream>>>(x, Wk, Kp, L_SEQ, NKV * HD, D_MODEL);
  gemm_xwT<<<dim3((NKV * HD) / BN, L_SEQ / BM), gblk, 0, stream>>>(x, Wv, Vp, L_SEQ, NKV * HD, D_MODEL);

  const int nq = L_SEQ * NH * (HD / 2);
  rope_cvt<<<(nq + 255) / 256, 256, 0, stream>>>(Qp, Qh, asc, NH, 0.125f);   // 0.125 = 64^-0.5
  const int nk = L_SEQ * NKV * (HD / 2);
  rope_cvt<<<(nk + 255) / 256, 256, 0, stream>>>(Kp, Kh, nullptr, NKV, 1.0f);
  const int nv = L_SEQ * NKV * HD;
  cvt_v<<<(nv + 255) / 256, 256, 0, stream>>>(Vp, Vh);

  attn_fwd<<<dim3(L_SEQ / 16, NH), dim3(32), 0, stream>>>(Qh, Kh, Vh, Ob);

  gemm_xwT<<<dim3(D_MODEL / BN, L_SEQ / BM), gblk, 0, stream>>>(Ob, Wo, out, L_SEQ, D_MODEL, D_MODEL);
}
